// SelfAttention_2113123910085
// MI455X (gfx1250) — matrix operand order repaired and checked by reading, not compiled, not run on hardware
//
#include <hip/hip_runtime.h>
#include <hip/hip_bf16.h>

typedef __attribute__((ext_vector_type(16))) __bf16 v16bf;
typedef __attribute__((ext_vector_type(8)))  __bf16 v8bf;
typedef __attribute__((ext_vector_type(8)))  float  v8f;

#define N_EMBD 1024
#define N_HEAD 16
#define HDIM   64
#define SEQ    2048
#define BATCH  4

__device__ __forceinline__ __bf16 f2bf(float x) {
  unsigned u = __builtin_bit_cast(unsigned, x);
  u += 0x7FFFu + ((u >> 16) & 1u);
  unsigned short hs = (unsigned short)(u >> 16);
  return __builtin_bit_cast(__bf16, hs);
}

union V16 { v16bf v; v8bf h[2]; };

// 16-byte async copy: global memory -> LDS, tracked by ASYNCcnt (gfx1250).
// Low 32 bits of a generic shared-memory address are the LDS byte offset
// (ISA 10.2).
__device__ __forceinline__ void async_ld16(void* lptr, const void* gptr) {
  unsigned l = (unsigned)(unsigned long long)(uintptr_t)lptr;
  unsigned long long g = (unsigned long long)(uintptr_t)gptr;
  asm volatile("global_load_async_to_lds_b128 %0, %1, off"
               :: "v"(l), "v"(g) : "memory");
}
__device__ __forceinline__ void wait_async0() {
  asm volatile("s_wait_asynccnt 0" ::: "memory");
}

// ---------------------------------------------------------------- fp32 -> bf16
__global__ void cvt_f32_bf16(const float* __restrict__ in, __bf16* __restrict__ out,
                             long long n) {
  long long i = (long long)blockIdx.x * blockDim.x + threadIdx.x;
  long long stride = (long long)gridDim.x * blockDim.x;
  for (; i < n; i += stride) out[i] = f2bf(in[i]);
}

// ------------------------------------------------- bf16 GEMM, f32 accumulate
// C[M,N] = A[M,K] * B[K,N] + bias[N]; block tile 128x128, 4 waves of 64x64.
template <bool STORE_F32>
__global__ __launch_bounds__(128)
void gemm_bf16_wmma(const __bf16* __restrict__ A, const __bf16* __restrict__ B,
                    const float* __restrict__ bias,
                    __bf16* __restrict__ Cb, float* __restrict__ Cf,
                    int M, int N, int K) {
  __shared__ alignas(32) __bf16 Alds[128][48];   // [m][k], stride 96B
  __shared__ alignas(32) __bf16 Blds[128][48];   // transposed: [n][k]

  const int tid  = threadIdx.x;
  const int lane = tid & 31;
  const int w    = tid >> 5;              // wave 0..3 (wave32)
  const int wm   = (w >> 1) * 64;
  const int wn   = (w & 1) * 64;
  const int m0   = blockIdx.y * 128;
  const int n0   = blockIdx.x * 128;
  const int hs   = (lane & 16) ? 1 : 0;   // lane-half select
  const int ln   = lane & 15;

  v8f acc[4][4];
  const v8f vzero = {};
#pragma unroll
  for (int i = 0; i < 4; ++i)
#pragma unroll
    for (int j = 0; j < 4; ++j) acc[i][j] = vzero;

  const int kkB = tid & 31;
  const int ccB = (tid >> 5) * 32;

  for (int k0 = 0; k0 < K; k0 += 32) {
    // --- stage A tile 128x32 via async global->LDS (ASYNCcnt path) ---
    {
      const __bf16* ap = A + (size_t)(m0 + tid) * K + k0;
#pragma unroll
      for (int j = 0; j < 4; ++j)
        async_ld16(&Alds[tid][j * 8], ap + j * 8);
      if (k0 + 32 < K) __builtin_prefetch(ap + 32, 0, 3);
    }
    // --- stage B tile 32x128 transposed (VGPR transpose) ---
    {
      const __bf16* bp = B + (size_t)(k0 + kkB) * N + n0 + ccB;
#pragma unroll
      for (int j = 0; j < 4; ++j) {
        v8bf vv = *(const v8bf*)(bp + j * 8);
#pragma unroll
        for (int e = 0; e < 8; ++e) Blds[ccB + j * 8 + e][kkB] = vv[e];
      }
      if (k0 + 32 < K) __builtin_prefetch(bp + (size_t)32 * N, 0, 3);
    }
    wait_async0();
    __syncthreads();

    // --- fragments (bf16 A 16x32 / B 32x16 lane layouts) ---
    V16 af[4], bfr[4];
    const int klo = hs * 16;
#pragma unroll
    for (int mi = 0; mi < 4; ++mi) {
      const int r = wm + mi * 16 + ln;
      af[mi].h[0] = *(const v8bf*)&Alds[r][klo];
      af[mi].h[1] = *(const v8bf*)&Alds[r][klo + 8];
    }
#pragma unroll
    for (int ni = 0; ni < 4; ++ni) {
      const int c = wn + ni * 16 + ln;
      bfr[ni].v = *(const v16bf*)&Blds[c][hs * 16];
    }
#pragma unroll
    for (int mi = 0; mi < 4; ++mi)
#pragma unroll
      for (int ni = 0; ni < 4; ++ni)
        acc[mi][ni] = __builtin_amdgcn_wmma_f32_16x16x32_bf16(
            false, af[mi].v, false, bfr[ni].v, (short)0, acc[mi][ni], false, false);
    __syncthreads();
  }

  // --- epilogue: + bias, store ---
#pragma unroll
  for (int mi = 0; mi < 4; ++mi) {
#pragma unroll
    for (int ni = 0; ni < 4; ++ni) {
      const int gc = n0 + wn + ni * 16 + ln;
      const float bv = bias[gc];
#pragma unroll
      for (int rr = 0; rr < 8; ++rr) {
        const int gr = m0 + wm + mi * 16 + rr + hs * 8;
        const float val = acc[mi][ni][rr] + bv;
        if (STORE_F32) Cf[(size_t)gr * N + gc] = val;
        else           Cb[(size_t)gr * N + gc] = f2bf(val);
      }
    }
  }
}

// --------------------------------------------- causal flash attention (bf16)
// grid: (T/64, B*H); block: 128 threads (4 waves x 16 q-rows).
__global__ __launch_bounds__(128)
void attn_wmma(const __bf16* __restrict__ QKV, __bf16* __restrict__ Y) {
  __shared__ alignas(32) __bf16 Klds[64][80];      // [t][d]
  __shared__ alignas(32) __bf16 Vt[64][80];        // [d][t] (transposed)
  __shared__ alignas(32) __bf16 Plds[4][16][80];   // per-wave P tile [m][t]

  const int tid  = threadIdx.x;
  const int lane = tid & 31;
  const int w    = tid >> 5;
  const int ln   = lane & 15;
  const int hs   = (lane & 16) ? 1 : 0;

  const int bh  = blockIdx.y;
  const int b   = bh >> 4;
  const int h   = bh & 15;
  const int qt0 = blockIdx.x * 64;
  const float scale = 0.125f;  // 1/sqrt(64)

  // Q fragments: rows qt0 + w*16 + (0..15), d split per A layout
  V16 qf[2];
  {
    const int tq = qt0 + w * 16 + ln;
    const __bf16* qp = QKV + ((size_t)b * SEQ + tq) * 3072 + h * 64;
    const int klo = hs * 16;
#pragma unroll
    for (int ks = 0; ks < 2; ++ks) {
      qf[ks].h[0] = *(const v8bf*)(qp + ks * 32 + klo);
      qf[ks].h[1] = *(const v8bf*)(qp + ks * 32 + klo + 8);
    }
  }

  float mrow[8], lrow[8];
  v8f o[4];
  const v8f vzero = {};
#pragma unroll
  for (int r = 0; r < 8; ++r) { mrow[r] = -3.0e38f; lrow[r] = 0.0f; }
#pragma unroll
  for (int d = 0; d < 4; ++d) o[d] = vzero;

  const int nkt = (qt0 / 64) + 1;  // causal k-tile bound
  for (int kt = 0; kt < nkt; ++kt) {
    const int kt0 = kt * 64;
    // --- K tile: async global->LDS; V tile: VGPR transpose ---
    {
      const int t  = tid >> 1;
      const int dh = (tid & 1) * 32;
      const __bf16* kp = QKV + ((size_t)b * SEQ + kt0 + t) * 3072 + 1024 + h * 64 + dh;
      const __bf16* vp = kp + 1024;
#pragma unroll
      for (int j = 0; j < 4; ++j)
        async_ld16(&Klds[t][dh + j * 8], kp + j * 8);
#pragma unroll
      for (int j = 0; j < 4; ++j) {
        v8bf vv = *(const v8bf*)(vp + j * 8);
#pragma unroll
        for (int e = 0; e < 8; ++e) Vt[dh + j * 8 + e][t] = vv[e];
      }
    }
    wait_async0();
    __syncthreads();

    // --- S = Q K^T ---
    v8f s[4];
#pragma unroll
    for (int nt = 0; nt < 4; ++nt) {
      s[nt] = vzero;
      const int tl = nt * 16 + ln;
      const int kb = hs * 16;
#pragma unroll
      for (int ks = 0; ks < 2; ++ks) {
        v16bf bk = *(const v16bf*)&Klds[tl][ks * 32 + kb];
        s[nt] = __builtin_amdgcn_wmma_f32_16x16x32_bf16(
            false, qf[ks].v, false, bk, (short)0, s[nt], false, false);
      }
    }

    // --- scale + causal mask + row max ---
    float mloc[8];
#pragma unroll
    for (int r = 0; r < 8; ++r) mloc[r] = -3.0e38f;
#pragma unroll
    for (int nt = 0; nt < 4; ++nt) {
      const int tk = kt0 + nt * 16 + ln;
#pragma unroll
      for (int r = 0; r < 8; ++r) {
        const int tq = qt0 + w * 16 + r + hs * 8;
        float v = s[nt][r] * scale;
        if (tk > tq) v = -3.0e38f;
        s[nt][r] = v;
        mloc[r] = fmaxf(mloc[r], v);
      }
    }
#pragma unroll
    for (int off = 1; off < 16; off <<= 1)
#pragma unroll
      for (int r = 0; r < 8; ++r)
        mloc[r] = fmaxf(mloc[r], __shfl_xor(mloc[r], off, 32));

    float alpha[8], psum[8];
#pragma unroll
    for (int r = 0; r < 8; ++r) {
      const float mnew = fmaxf(mrow[r], mloc[r]);
      alpha[r] = __expf(mrow[r] - mnew);
      mrow[r] = mnew;
      psum[r] = 0.0f;
    }
    // --- exponentiate, spill P (C-layout -> LDS row-major) ---
#pragma unroll
    for (int nt = 0; nt < 4; ++nt) {
#pragma unroll
      for (int r = 0; r < 8; ++r) {
        const float p = __expf(s[nt][r] - mrow[r]);
        psum[r] += p;
        Plds[w][r + hs * 8][nt * 16 + ln] = f2bf(p);
      }
    }
#pragma unroll
    for (int off = 1; off < 16; off <<= 1)
#pragma unroll
      for (int r = 0; r < 8; ++r)
        psum[r] += __shfl_xor(psum[r], off, 32);
#pragma unroll
    for (int r = 0; r < 8; ++r) lrow[r] = lrow[r] * alpha[r] + psum[r];
#pragma unroll
    for (int d = 0; d < 4; ++d)
#pragma unroll
      for (int r = 0; r < 8; ++r) o[d][r] *= alpha[r];

    __syncthreads();

    // --- O += P V ---
    V16 pf[2];
    {
      const int klo = hs * 16;
#pragma unroll
      for (int ks = 0; ks < 2; ++ks) {
        pf[ks].h[0] = *(const v8bf*)&Plds[w][ln][ks * 32 + klo];
        pf[ks].h[1] = *(const v8bf*)&Plds[w][ln][ks * 32 + klo + 8];
      }
    }
#pragma unroll
    for (int dt = 0; dt < 4; ++dt) {
      const int dl = dt * 16 + ln;
      const int kb = hs * 16;
#pragma unroll
      for (int ks = 0; ks < 2; ++ks) {
        v16bf bv = *(const v16bf*)&Vt[dl][ks * 32 + kb];
        o[dt] = __builtin_amdgcn_wmma_f32_16x16x32_bf16(
            false, pf[ks].v, false, bv, (short)0, o[dt], false, false);
      }
    }
    __syncthreads();
  }

  // --- epilogue: O /= l, store bf16 y[b,t,h*64+d] ---
#pragma unroll
  for (int dt = 0; dt < 4; ++dt) {
    const int d = dt * 16 + ln;
#pragma unroll
    for (int r = 0; r < 8; ++r) {
      const int tq = qt0 + w * 16 + r + hs * 8;
      const float val = o[dt][r] / lrow[r];
      Y[((size_t)b * SEQ + tq) * N_EMBD + h * HDIM + d] = f2bf(val);
    }
  }
}

// ------------------------------------------------------------------ launcher
extern "C" void kernel_launch(void* const* d_in, const int* in_sizes, int n_in,
                              void* d_out, int out_size, void* d_ws, size_t ws_size,
                              hipStream_t stream) {
  (void)in_sizes; (void)n_in; (void)out_size; (void)ws_size;
  const float* x      = (const float*)d_in[0];
  const float* W_attn = (const float*)d_in[1];
  const float* b_attn = (const float*)d_in[2];
  const float* W_proj = (const float*)d_in[3];
  const float* b_proj = (const float*)d_in[4];
  float* out = (float*)d_out;

  const size_t BT = (size_t)BATCH * SEQ;  // 8192
  char* ws = (char*)d_ws;
  __bf16* xb   = (__bf16*)ws; ws += BT * N_EMBD * 2;
  __bf16* Wab  = (__bf16*)ws; ws += (size_t)N_EMBD * 3 * N_EMBD * 2;
  __bf16* Wpb  = (__bf16*)ws; ws += (size_t)N_EMBD * N_EMBD * 2;
  __bf16* QKVb = (__bf16*)ws; ws += BT * 3 * N_EMBD * 2;
  __bf16* Yb   = (__bf16*)ws; ws += BT * N_EMBD * 2;

  cvt_f32_bf16<<<512, 256, 0, stream>>>(x, xb, (long long)(BT * N_EMBD));
  cvt_f32_bf16<<<512, 256, 0, stream>>>(W_attn, Wab, (long long)N_EMBD * 3 * N_EMBD);
  cvt_f32_bf16<<<512, 256, 0, stream>>>(W_proj, Wpb, (long long)N_EMBD * N_EMBD);

  gemm_bf16_wmma<false><<<dim3(3 * N_EMBD / 128, (int)(BT / 128)), 128, 0, stream>>>(
      xb, Wab, b_attn, QKVb, nullptr, (int)BT, 3 * N_EMBD, N_EMBD);

  attn_wmma<<<dim3(SEQ / 64, BATCH * N_HEAD), 128, 0, stream>>>(QKVb, Yb);

  gemm_bf16_wmma<true><<<dim3(N_EMBD / 128, (int)(BT / 128)), 128, 0, stream>>>(
      Yb, Wpb, b_proj, nullptr, out, (int)BT, N_EMBD, N_EMBD);
}
